// DenseOnUp_23562190586024
// MI455X (gfx1250) — matrix operand order repaired and checked by reading, not compiled, not run on hardware
//
#include <hip/hip_runtime.h>
#include <hip/hip_bf16.h>

typedef __attribute__((ext_vector_type(16))) __bf16 v16bf;
typedef __attribute__((ext_vector_type(8)))  __bf16 v8bf;
typedef __attribute__((ext_vector_type(8)))  float  v8f;

#define N_ROWS 400000
#define F_DIM  128
#define M_SEL  100000
#define D0     256
#define D1     256
#define D2     128

// ---- LDS layout (bytes). Weights stored transposed: Wt[n][k], bf16,
// row strides padded so stride(dwords) % 64 == 4 -> conflict-free B loads.
constexpr int W0T_S = 136;  // K stride for W0t [256][136], valid k<128
constexpr int W1T_S = 264;  // W1t [256][264], valid k<256
constexpr int W2T_S = 264;  // W2t [128][264], valid k<256
constexpr int SCR_S = 272;  // per-wave activation scratch [16][272] bf16

constexpr int W0T_OFF = 0;
constexpr int W1T_OFF = W0T_OFF + 256 * W0T_S * 2;          // 69632
constexpr int W2T_OFF = W1T_OFF + 256 * W1T_S * 2;          // 204800
constexpr int SCR_OFF = W2T_OFF + 128 * W2T_S * 2;          // 272384
constexpr int SMEM_BYTES = SCR_OFF + 4 * 16 * SCR_S * 2;    // 307200 <= 320KB

// ---- A fragment: 16x32 bf16, ISA 7.12.2 layout.
// lane<16: row=lane, elems 0..7 -> K=k0..k0+7, elems 8..15 -> K=k0+16..k0+23 (khalf=0)
// lane>=16: same row, K offset by +8 within each half.
__device__ __forceinline__ v16bf load_A(const __bf16* base, int S, int lane, int kstep) {
  int row   = lane & 15;
  int khalf = lane >> 4;
  const __bf16* p = base + row * S + kstep * 32 + khalf * 8;
  v8bf lo = *(const v8bf*)(p);        // K = kstep*32 + khalf*8 + 0..7
  v8bf hi = *(const v8bf*)(p + 16);   // K = kstep*32 + 16 + khalf*8 + 0..7
  return __builtin_shufflevector(lo, hi, 0,1,2,3,4,5,6,7,8,9,10,11,12,13,14,15);
}

// ---- B fragment: 32x16 bf16. lane<16: K=0..7,16..23; lane>=16: K=8..15,24..31; N = lane&15.
// Weights stored as Wt[n][k] so each lane's K-values form two contiguous runs of 8.
__device__ __forceinline__ v16bf load_B(const __bf16* wt, int S, int lane, int ntile, int kstep) {
  int col = ntile * 16 + (lane & 15);
  int kb  = kstep * 32 + (lane >> 4) * 8;
  const __bf16* p = wt + col * S + kb;
  v8bf lo = *(const v8bf*)(p);
  v8bf hi = *(const v8bf*)(p + 16);
  return __builtin_shufflevector(lo, hi, 0,1,2,3,4,5,6,7,8,9,10,11,12,13,14,15);
}

__device__ __forceinline__ float elu(float r) {
  return r > 0.0f ? r : (__expf(r) - 1.0f);
}

// One 16-row x (NTILES*16)-col matmul, K = KSTEPS*32. Keeps all N-tile
// accumulators live so each A fragment is loaded once per K-step.
template <int KSTEPS, int NTILES>
__device__ __forceinline__ void mm_tiles(const __bf16* actIn, int inS,
                                         const __bf16* wt, int wtS,
                                         int lane, v8f* acc) {
#pragma unroll
  for (int n = 0; n < NTILES; ++n)
    acc[n] = (v8f){0.f, 0.f, 0.f, 0.f, 0.f, 0.f, 0.f, 0.f};
  for (int k = 0; k < KSTEPS; ++k) {
    v16bf a = load_A(actIn, inS, lane, k);
#pragma unroll
    for (int n = 0; n < NTILES; ++n) {
      v16bf b = load_B(wt, wtS, lane, n, k);
      acc[n] = __builtin_amdgcn_wmma_f32_16x16x32_bf16(
          /*neg_a=*/false, a, /*neg_b=*/false, b,
          /*c_mod=*/(short)0, acc[n], /*reuse_a=*/false, /*reuse_b=*/false);
    }
  }
}

// Bias + ELU + bf16 store back to scratch. D layout: VGPR v -> M = v + (lane>>4)*8,
// N = lane&15 (same column for all 8 elements of a lane).
template <int NTILES>
__device__ __forceinline__ void epilogue_store(const v8f* acc, const float* __restrict__ bias,
                                               __bf16* actOut, int outS, int lane) {
  int nc = lane & 15;
  int mbase = (lane >> 4) * 8;
#pragma unroll
  for (int n = 0; n < NTILES; ++n) {
    int col = n * 16 + nc;
    float bv = bias[col];
#pragma unroll
    for (int v = 0; v < 8; ++v) {
      float r = elu(acc[n][v] + bv);
      actOut[(mbase + v) * outS + col] = (__bf16)r;
    }
  }
}

__global__ __launch_bounds__(128, 1) void fused_mlp_scatter(
    const float* __restrict__ x, const int* __restrict__ sel,
    const float* __restrict__ W0, const float* __restrict__ b0,
    const float* __restrict__ W1, const float* __restrict__ b1,
    const float* __restrict__ W2, const float* __restrict__ b2,
    float* __restrict__ out) {
  extern __shared__ __align__(16) char smem[];
  __bf16* w0t = (__bf16*)(smem + W0T_OFF);
  __bf16* w1t = (__bf16*)(smem + W1T_OFF);
  __bf16* w2t = (__bf16*)(smem + W2T_OFF);

  const int tid  = threadIdx.x;
  const int lane = tid & 31;
  const int wave = tid >> 5;

  // ---- Stage weights into LDS, transposed [n][k], f32 -> bf16.
  // Consecutive threads read consecutive n -> coalesced global loads.
  for (int i = tid; i < F_DIM * D0; i += 128) {
    int k = i / D0, n = i % D0;
    w0t[n * W0T_S + k] = (__bf16)W0[i];
  }
  for (int i = tid; i < D0 * D1; i += 128) {
    int k = i / D1, n = i % D1;
    w1t[n * W1T_S + k] = (__bf16)W1[i];
  }
  for (int i = tid; i < D1 * D2; i += 128) {
    int k = i / D2, n = i % D2;
    w2t[n * W2T_S + k] = (__bf16)W2[i];
  }
  __syncthreads();

  __bf16* scr = (__bf16*)(smem + SCR_OFF) + wave * (16 * SCR_S);

  const int nTiles = M_SEL / 16;  // 6250 (M is a multiple of 16)
  for (int t = blockIdx.x * 4 + wave; t < nTiles; t += gridDim.x * 4) {
    const int tbase = t * 16;

    // ---- Gather 16 selected rows of x into per-wave scratch as bf16.
    {
      int row = lane >> 1, half = lane & 1;
      int gr = sel[tbase + row];
      const float4* src = (const float4*)(x + (size_t)gr * F_DIM + half * 64);
      __bf16* dst = scr + row * SCR_S + half * 64;
#pragma unroll
      for (int i = 0; i < 8; ++i) {
        float4 f0 = src[2 * i];
        float4 f1 = src[2 * i + 1];
        v8bf o = {(__bf16)f0.x, (__bf16)f0.y, (__bf16)f0.z, (__bf16)f0.w,
                  (__bf16)f1.x, (__bf16)f1.y, (__bf16)f1.z, (__bf16)f1.w};
        *(v8bf*)(dst + i * 8) = o;
      }
    }

    // Global row indices for the 8 M-rows this lane's accumulators map to.
    const int mbase = (lane >> 4) * 8;
    int gidx[8];
#pragma unroll
    for (int v = 0; v < 8; ++v) gidx[v] = sel[tbase + mbase + v];

    // ---- Layer 0: [16x128] @ [128x256]
    {
      v8f acc[16];
      mm_tiles<4, 16>(scr, SCR_S, w0t, W0T_S, lane, acc);
      epilogue_store<16>(acc, b0, scr, SCR_S, lane);
    }
    // ---- Layer 1: [16x256] @ [256x256]
    {
      v8f acc[16];
      mm_tiles<8, 16>(scr, SCR_S, w1t, W1T_S, lane, acc);
      epilogue_store<16>(acc, b1, scr, SCR_S, lane);
    }
    // ---- Layer 2: [16x256] @ [256x128] + ELU + scatter-add to out.
    {
      v8f acc[8];
      mm_tiles<8, 8>(scr, SCR_S, w2t, W2T_S, lane, acc);
      int nc = lane & 15;
#pragma unroll
      for (int n = 0; n < 8; ++n) {
        int col = n * 16 + nc;
        float bv = b2[col];
#pragma unroll
        for (int v = 0; v < 8; ++v) {
          float r = elu(acc[n][v] + bv);
          __hip_atomic_fetch_add(&out[(size_t)gidx[v] * F_DIM + col], r,
                                 __ATOMIC_RELAXED, __HIP_MEMORY_SCOPE_AGENT);
        }
      }
    }
  }
}

extern "C" void kernel_launch(void* const* d_in, const int* in_sizes, int n_in,
                              void* d_out, int out_size, void* d_ws, size_t ws_size,
                              hipStream_t stream) {
  (void)in_sizes; (void)n_in; (void)out_size; (void)d_ws; (void)ws_size;
  const float* x   = (const float*)d_in[0];
  const int*   sel = (const int*)d_in[1];
  const float* W0  = (const float*)d_in[2];
  const float* b0  = (const float*)d_in[3];
  const float* W1  = (const float*)d_in[4];
  const float* b1  = (const float*)d_in[5];
  const float* W2  = (const float*)d_in[6];
  const float* b2  = (const float*)d_in[7];
  float* out = (float*)d_out;

  // out = x (the scatter kernel then atomically adds the MLP results).
  hipMemcpyAsync(out, x, sizeof(float) * (size_t)N_ROWS * F_DIM,
                 hipMemcpyDeviceToDevice, stream);

  dim3 grid(512), block(128);
  hipLaunchKernelGGL(fused_mlp_scatter, grid, block, SMEM_BYTES, stream,
                     x, sel, W0, b0, W1, b1, W2, b2, out);
}
